// FourierKANLayer_64132451664281
// MI455X (gfx1250) — compile-verified, matrix operand order checked by reading, not run on hardware
//
#include <hip/hip_runtime.h>
#include <hip/hip_bf16.h>

#define IN_F   128
#define OUT_F  128
#define NMODES 32
#define KTOT   (IN_F * 2 * NMODES)      // 8192 (bias column folded out)
#define NKC    (KTOT / 32)              // 256 k-chunks of 32
#define NNT    (OUT_F / 16)             // 8 n-tiles
#define WG_ROWS 128                     // rows per workgroup (8 waves x 16)
#define ROW_STRIDE_B 136                // 64 bf16 (=128B) + 8B pad
#define BUF_BYTES (WG_ROWS * ROW_STRIDE_B)   // 17408
#define B_BYTES (KTOT * OUT_F * 2)      // 2 MB packed bf16 B matrix

typedef __attribute__((ext_vector_type(16))) __bf16 v16bf;
typedef __attribute__((ext_vector_type(8)))  float  v8f;

__device__ __forceinline__ unsigned short f2bf(float f) {
    unsigned int u = __float_as_uint(f);
    u = (u + 0x7FFFu + ((u >> 16) & 1u)) >> 16;   // RNE
    return (unsigned short)u;
}

// Pack two f32 -> two bf16 (round-to-nearest) in 3 VALU: 2x add + 1x v_perm_b32
__device__ __forceinline__ unsigned int pkbf(float lo, float hi) {
#if __has_builtin(__builtin_amdgcn_perm)
    unsigned int ul = __float_as_uint(lo) + 0x8000u;
    unsigned int uh = __float_as_uint(hi) + 0x8000u;
    return __builtin_amdgcn_perm(uh, ul, 0x07060302u); // [uh.b3 uh.b2 ul.b3 ul.b2]
#else
    return (unsigned int)f2bf(lo) | ((unsigned int)f2bf(hi) << 16);
#endif
}

// ---------------------------------------------------------------------------
// Prep 1: repack coeffs (f32 [o][i][65]) -> bf16 B in WMMA fragment layout.
// Block b = kc*8+nt is 1024B: lane l owns bytes [l*32, l*32+32):
//   bytes 0..15  = VGPR0-3 (low K set), 16..31 = VGPR4-7 (high K set)
//   lanes 0-15:  N=nt*16+l,      K-in-chunk {0..7, 16..23}
//   lanes 16-31: N=nt*16+(l-16), K-in-chunk {8..15,24..31}
// K global = kc*32+k ; i = K/64 ; r = K%64 ; b = r<32 ? 1+r : 33+(r-32)
// ---------------------------------------------------------------------------
__global__ __launch_bounds__(256) void fkan_prep_b(const float* __restrict__ coeffs,
                                                   unsigned char* __restrict__ Bp) {
    int t = blockIdx.x * 256 + threadIdx.x;          // 0..65535
    int lane = t & 31;
    unsigned int dw[8];
#pragma unroll
    for (int j = 0; j < 16; ++j) {
        int k = (lane < 16) ? ((j < 8) ? j : 8 + j)          // 0..7 , 16..23
                            : ((j < 8) ? 8 + j : 16 + j);    // 8..15, 24..31
        int K = ((t >> 8) << 5) + k;                 // kc*32 + k
        int i = K >> 6;
        int r = K & 63;
        int b = (r < 32) ? (1 + r) : (33 + (r - 32));
        int nt = (t >> 5) & 7;
        int n  = nt * 16 + (lane & 15);
        float v = coeffs[(n * IN_F + i) * 65 + b];
        unsigned short h = f2bf(v);
        if (j & 1) dw[j >> 1] |= (unsigned int)h << 16;
        else       dw[j >> 1]  = h;
    }
    uint4* dst = (uint4*)(Bp) + (size_t)t * 2;       // t*32 bytes
    dst[0] = make_uint4(dw[0], dw[1], dw[2], dw[3]);
    dst[1] = make_uint4(dw[4], dw[5], dw[6], dw[7]);
}

// Prep 2: bias[o] = sum_i coeffs[o][i][0]
__global__ __launch_bounds__(128) void fkan_bias(const float* __restrict__ coeffs,
                                                 float* __restrict__ bias) {
    int o = threadIdx.x;
    float s = 0.f;
    for (int i = 0; i < IN_F; ++i) s += coeffs[(o * IN_F + i) * 65];
    bias[o] = s;
}

// ---------------------------------------------------------------------------
// Basis staging: 2 threads per row, split by MODE RANGE (each computes both
// sin & cos for 16 modes). part p covers k = p*16+1 .. p*16+16; k=17 start
// built branchlessly from 4 complex squarings + one multiply.
// ---------------------------------------------------------------------------
__device__ __forceinline__ void stage_basis(const float* __restrict__ x,
                                            unsigned char* __restrict__ smem,
                                            int row0, int srow, int part,
                                            int i, int wbuf) {
    const float PI = 3.14159265358979323846f;
    float xv = x[(size_t)(row0 + srow) * IN_F + i];
    float th = PI * xv;
    float s1, c1;
    __sincosf(th, &s1, &c1);
    // e^{i*17*th} via squarings: ^2 ^4 ^8 ^16, then * e^{i th}
    float s2  = 2.f * s1 * c1,  c2  = fmaf(c1, c1, -s1 * s1);
    float s4  = 2.f * s2 * c2,  c4  = fmaf(c2, c2, -s2 * s2);
    float s8  = 2.f * s4 * c4,  c8  = fmaf(c4, c4, -s4 * s4);
    float s16 = 2.f * s8 * c8,  c16 = fmaf(c8, c8, -s8 * s8);
    float s17 = fmaf(s16, c1,  c16 * s1);
    float c17 = fmaf(c16, c1, -s16 * s1);
    float sk = part ? s17 : s1;
    float ck = part ? c17 : c1;
    unsigned char* wp = smem + wbuf * BUF_BYTES + srow * ROW_STRIDE_B + part * 32;
#pragma unroll
    for (int k = 0; k < 16; k += 4) {
        float ss[4], cc[4];
#pragma unroll
        for (int j = 0; j < 4; ++j) {
            ss[j] = sk; cc[j] = ck;
            float sn = fmaf(sk, c1,  ck * s1);
            float cn = fmaf(ck, c1, -sk * s1);
            sk = sn; ck = cn;
        }
        *(uint2*)(wp + k * 2)      = make_uint2(pkbf(ss[0], ss[1]), pkbf(ss[2], ss[3]));
        *(uint2*)(wp + 64 + k * 2) = make_uint2(pkbf(cc[0], cc[1]), pkbf(cc[2], cc[3]));
    }
}

// ---------------------------------------------------------------------------
// Main: per workgroup, 128 rows x 128 outputs. Body order per iteration:
//   1) ds_load A fragments (ahead of staging stores in the in-order DS queue,
//      so WMMA waits only on the loads, not on next-buffer stores)
//   2) stage(i+1) into the other LDS buffer (VALU hidden under WMMA stream)
//   3) 16x v_wmma_f32_16x16x32_bf16 with clause-grouped B loads from L2
// ---------------------------------------------------------------------------
__global__ __launch_bounds__(256, 1)
void fkan_main(const float* __restrict__ x,
               const uint4* __restrict__ Bp,
               const float* __restrict__ bias,
               const float* __restrict__ scale_p,
               float* __restrict__ out) {
    __shared__ __align__(16) unsigned char smem[2 * BUF_BYTES];

    const int tid  = threadIdx.x;
    const int wave = tid >> 5;
    const int lane = tid & 31;
    const int row0 = blockIdx.x * WG_ROWS;
    const float scale = *scale_p;

    v8f acc[NNT];
#pragma unroll
    for (int nt = 0; nt < NNT; ++nt) acc[nt] = (v8f)0.f;

    // A-fragment LDS read geometry
    const int mlocal = wave * 16 + (lane & 15);
    const int klo    = (lane < 16) ? 0 : 8;
    const unsigned int abase0 = (unsigned int)(mlocal * ROW_STRIDE_B + klo * 2);

    // staging geometry: 2 threads per row, mode-range split
    const int srow = tid >> 1;
    const int part = tid & 1;

    stage_basis(x, smem, row0, srow, part, 0, 0);
    __syncthreads();

    for (int i = 0; i < IN_F; ++i) {
        const int buf = i & 1;

        // ---- 1) A fragments for both k-chunks, issued before staging stores
        union { v16bf v; uint2 q[4]; } A[2];
#pragma unroll
        for (int h = 0; h < 2; ++h) {
            const unsigned char* rp = smem + buf * BUF_BYTES + abase0 + h * 64;
            A[h].q[0] = *(const uint2*)(rp);        // K = klo..klo+3
            A[h].q[1] = *(const uint2*)(rp + 8);    // K = klo+4..klo+7
            A[h].q[2] = *(const uint2*)(rp + 32);   // K = klo+16..klo+19
            A[h].q[3] = *(const uint2*)(rp + 40);   // K = klo+20..klo+23
        }

        // ---- 2) stage next feature into the other buffer (clamped at end;
        //          final redundant stage lands in the dead buffer)
        const int is = (i < IN_F - 1) ? (i + 1) : (IN_F - 1);
        stage_basis(x, smem, row0, srow, part, is, (i + 1) & 1);

        // ---- 3) consume feature i: 2 k-chunks x 8 n-tiles of WMMA
#pragma unroll
        for (int h = 0; h < 2; ++h) {
            const int kc = i * 2 + h;
            const uint4* bptr = Bp + ((size_t)kc * NNT) * 64 + lane * 2;
#pragma unroll
            for (int nt = 0; nt < NNT; ++nt) {
                union { v16bf v; uint4 q[2]; } B;
                B.q[0] = bptr[0];
                B.q[1] = bptr[1];
                bptr += 64;                      // next 1KB fragment block
                acc[nt] = __builtin_amdgcn_wmma_f32_16x16x32_bf16(
                    false, A[h].v, false, B.v, (short)0, acc[nt], false, false);
            }
        }
        __syncthreads();
    }

    // ---- epilogue: D layout: VGPR j -> M = j + (lane<16 ? 0 : 8), N = lane%16
    const int mbase = row0 + wave * 16 + ((lane < 16) ? 0 : 8);
    const int ncol  = lane & 15;
#pragma unroll
    for (int nt = 0; nt < NNT; ++nt) {
        const int col = nt * 16 + ncol;
        const float bo = bias[col];
#pragma unroll
        for (int j = 0; j < 8; ++j) {
            const size_t idx = (size_t)(mbase + j) * OUT_F + col;
            out[idx] = acc[nt][j] + bo + scale * x[idx];
        }
    }
}

extern "C" void kernel_launch(void* const* d_in, const int* in_sizes, int n_in,
                              void* d_out, int out_size, void* d_ws, size_t ws_size,
                              hipStream_t stream) {
    const float* x      = (const float*)d_in[0];
    const float* coeffs = (const float*)d_in[1];
    const float* rscale = (const float*)d_in[2];
    float* out = (float*)d_out;

    unsigned char* Bp   = (unsigned char*)d_ws;          // 2 MB packed bf16 B
    float*         bias = (float*)(Bp + B_BYTES);        // 128 f32

    const int nRows = in_sizes[0] / IN_F;                // 32768

    fkan_prep_b<<<(NKC * NNT * 32) / 256, 256, 0, stream>>>(coeffs, Bp);
    fkan_bias<<<1, 128, 0, stream>>>(coeffs, bias);
    fkan_main<<<nRows / WG_ROWS, 256, 0, stream>>>(x, (const uint4*)Bp, bias, rscale, out);
}
